// SAINTTransformer_75170517615112
// MI455X (gfx1250) — matrix operand order repaired and checked by reading, not compiled, not run on hardware
//
#include <hip/hip_runtime.h>
#include <hip/hip_bf16.h>

// ---------------- problem constants ----------------
#define BB   1024
#define HH   16
#define FF   24
#define DD   128
#define NHD  8
#define DH   16
#define SS   17          // H + 1 (cls token appended)
#define DFF  512
#define NTOK (BB * SS * FF)   // 417792, multiple of 32

typedef unsigned short u16;
typedef __attribute__((ext_vector_type(16))) u16    u16x16;
typedef __attribute__((ext_vector_type(8)))  u16    u16x8;
typedef __attribute__((ext_vector_type(16))) __bf16 v16bf;
typedef __attribute__((ext_vector_type(8)))  float  v8f;

union U16V { u16x16 u; v16bf b; };

// ---------------- helpers ----------------
__device__ __forceinline__ u16 f2bf(float f) {
    unsigned int u = __float_as_uint(f);
    u += 0x7FFFu + ((u >> 16) & 1u);   // round-to-nearest-even
    return (u16)(u >> 16);
}
__device__ __forceinline__ float bf2f(u16 h) {
    return __uint_as_float(((unsigned int)h) << 16);
}
__device__ __forceinline__ float gelu_tanh(float x) {
    const float c = 0.7978845608028654f;
    float t = tanhf(c * (x + 0.044715f * x * x * x));
    return 0.5f * x * (1.0f + t);
}

// ---------------- weight prep: fp32 KxN -> bf16 transposed NxK ----------------
__global__ __launch_bounds__(256) void wprep_kernel(const float* __restrict__ W,
                                                    u16* __restrict__ Wt,
                                                    int K, int N) {
    int idx = blockIdx.x * 256 + threadIdx.x;
    if (idx >= K * N) return;
    int k = idx / N, n = idx % N;
    Wt[(size_t)n * K + k] = f2bf(W[idx]);
}

// ---------------- concat x + cls -> h (B,S,F,D) fp32 ----------------
__global__ __launch_bounds__(256) void concat_kernel(const float* __restrict__ x,
                                                     const float* __restrict__ cls,
                                                     float* __restrict__ h) {
    int idx = blockIdx.x * 256 + threadIdx.x;          // over NTOK*DD, exact
    int d   = idx & (DD - 1);
    int t   = idx >> 7;
    int f   = t % FF;
    int bs  = t / FF;
    int s   = bs % SS;
    int b   = bs / SS;
    float v = (s < HH) ? x[(((size_t)(b * HH + s)) * FF + f) * DD + d]
                       : cls[f * DD + d];
    h[idx] = v;
}

// ---------------- LayerNorm: fp32 token rows -> bf16 rows ----------------
__global__ __launch_bounds__(256) void ln_kernel(const float* __restrict__ h,
                                                 const float* __restrict__ g,
                                                 const float* __restrict__ bta,
                                                 u16* __restrict__ out, int ntok) {
    int lane = threadIdx.x & 31;
    int tok  = blockIdx.x * 8 + (threadIdx.x >> 5);
    if (tok >= ntok) return;
    const float* row = h + (size_t)tok * DD;
    float x0[4], s = 0.f;
#pragma unroll
    for (int i = 0; i < 4; ++i) { x0[i] = row[lane + 32 * i]; s += x0[i]; }
#pragma unroll
    for (int o = 16; o > 0; o >>= 1) s += __shfl_xor(s, o, 32);
    float mu = s * (1.0f / DD);
    float vs = 0.f;
#pragma unroll
    for (int i = 0; i < 4; ++i) { float d = x0[i] - mu; vs += d * d; }
#pragma unroll
    for (int o = 16; o > 0; o >>= 1) vs += __shfl_xor(vs, o, 32);
    float rstd = rsqrtf(vs * (1.0f / DD) + 1e-5f);
    u16* orow = out + (size_t)tok * DD;
#pragma unroll
    for (int i = 0; i < 4; ++i) {
        int c = lane + 32 * i;
        orow[c] = f2bf((x0[i] - mu) * rstd * g[c] + bta[c]);
    }
}

// ---------------- generic WMMA bf16 GEMM, 16x32 tile per wave ----------------
// C(M x N) = A(M x K, bf16 row-major) * Wt(N x K, bf16)^T
// Block = 8 waves: 2 m-tiles x 4 n-tile-pairs -> 32 x 128 output per block.
// Each wave keeps TWO independent accumulators (n0, n0+16) and reuses its A
// fragment across both -> two interleaved WMMA chains, no D->C RAW stalls.
// mode 0: store bf16 (+bias)      mode 1: store bf16 gelu(acc+bias)
// mode 2: outF[m*N+n] += acc + bias   (residual accumulate, fp32)
__global__ __launch_bounds__(256) void gemm_wmma_kernel(const u16* __restrict__ A,
                                                        const u16* __restrict__ Wt,
                                                        const float* __restrict__ bias,
                                                        u16* __restrict__ outB,
                                                        float* __restrict__ outF,
                                                        int N, int K, int mode) {
    int lane = threadIdx.x & 31;
    int wave = threadIdx.x >> 5;
    int m0 = blockIdx.x * 32 + (wave >> 2) * 16;
    int n0 = blockIdx.y * 128 + (wave & 3) * 32;

    int row  = m0 + (lane & 15);
    int col0 = n0 + (lane & 15);
    const u16* aRow  = A  + (size_t)row  * K;
    const u16* bRow0 = Wt + (size_t)col0 * K;
    const u16* bRow1 = bRow0 + (size_t)16 * K;
    int kAoff = (lane >> 4) ? 16 : 0;   // A: lanes16-31 hold K=16..31
    int kBoff = (lane >> 4) ? 16 : 0;   // B: lanes16-31 hold K=16..31

    v8f acc0 = {};
    v8f acc1 = {};
    for (int k0 = 0; k0 < K; k0 += 32) {
        U16V ua, ub0, ub1;
        u16x8 a0 = *reinterpret_cast<const u16x8*>(aRow + k0 + kAoff);
        u16x8 a1 = *reinterpret_cast<const u16x8*>(aRow + k0 + kAoff + 8);
#pragma unroll
        for (int j = 0; j < 8; ++j) { ua.u[j] = a0[j]; ua.u[8 + j] = a1[j]; }
        ub0.u = *reinterpret_cast<const u16x16*>(bRow0 + k0 + kBoff);
        ub1.u = *reinterpret_cast<const u16x16*>(bRow1 + k0 + kBoff);
        acc0 = __builtin_amdgcn_wmma_f32_16x16x32_bf16(
            false, ua.b, false, ub0.b, (short)0, acc0, false, false);
        acc1 = __builtin_amdgcn_wmma_f32_16x16x32_bf16(
            false, ua.b, false, ub1.b, (short)0, acc1, false, false);
    }

    int nlo   = n0 + (lane & 15);
    int mbase = m0 + ((lane >> 4) ? 8 : 0);
    float bia0 = bias ? bias[nlo]      : 0.0f;
    float bia1 = bias ? bias[nlo + 16] : 0.0f;
    if (mode == 2) {
#pragma unroll
        for (int r = 0; r < 8; ++r) {
            size_t idx = (size_t)(mbase + r) * N + nlo;
            outF[idx]      += acc0[r] + bia0;
            outF[idx + 16] += acc1[r] + bia1;
        }
    } else if (mode == 1) {
#pragma unroll
        for (int r = 0; r < 8; ++r) {
            size_t idx = (size_t)(mbase + r) * N + nlo;
            outB[idx]      = f2bf(gelu_tanh(acc0[r] + bia0));
            outB[idx + 16] = f2bf(gelu_tanh(acc1[r] + bia1));
        }
    } else {
#pragma unroll
        for (int r = 0; r < 8; ++r) {
            size_t idx = (size_t)(mbase + r) * N + nlo;
            outB[idx]      = f2bf(acc0[r] + bia0);
            outB[idx + 16] = f2bf(acc1[r] + bia1);
        }
    }
}

// ---------------- feature attention: per (b,s), wave = head, F=24 ----------------
__global__ __launch_bounds__(256) void feat_attn_kernel(const u16* __restrict__ q,
                                                        const u16* __restrict__ k,
                                                        const u16* __restrict__ v,
                                                        u16* __restrict__ o) {
    __shared__ float kst[NHD][FF][DH];
    __shared__ float vst[NHD][FF][DH];
    int lane = threadIdx.x & 31;
    int head = threadIdx.x >> 5;
    size_t t0 = (size_t)blockIdx.x * FF;   // first token of this (b,s)

    for (int i = lane; i < FF * DH; i += 32) {
        int g = i >> 4, e = i & 15;
        size_t idx = (t0 + g) * DD + head * DH + e;
        kst[head][g][e] = bf2f(k[idx]);
        vst[head][g][e] = bf2f(v[idx]);
    }
    __syncthreads();

    if (lane < FF) {
        int f = lane;
        const u16* qr = q + (t0 + f) * DD + head * DH;
        float qv[DH];
#pragma unroll
        for (int e = 0; e < DH; ++e) qv[e] = bf2f(qr[e]);
        float sc[FF], mx = -3.0e38f;
#pragma unroll
        for (int g = 0; g < FF; ++g) {
            float d = 0.f;
#pragma unroll
            for (int e = 0; e < DH; ++e) d += qv[e] * kst[head][g][e];
            d *= 0.25f;                    // 1/sqrt(16)
            sc[g] = d; mx = fmaxf(mx, d);
        }
        float sum = 0.f;
#pragma unroll
        for (int g = 0; g < FF; ++g) { sc[g] = __expf(sc[g] - mx); sum += sc[g]; }
        float inv = 1.0f / sum;
        u16* orow = o + (t0 + f) * DD + head * DH;
#pragma unroll
        for (int e = 0; e < DH; ++e) {
            float a = 0.f;
#pragma unroll
            for (int g = 0; g < FF; ++g) a += sc[g] * vst[head][g][e];
            orow[e] = f2bf(a * inv);
        }
    }
}

// ---------------- sequence attention: per (b,f), wave = head, S=17, masked ----------------
__global__ __launch_bounds__(256) void seq_attn_kernel(const u16* __restrict__ q,
                                                       const u16* __restrict__ k,
                                                       const u16* __restrict__ v,
                                                       const int* __restrict__ amask,
                                                       u16* __restrict__ o) {
    __shared__ float kst[NHD][SS][DH];
    __shared__ float vst[NHD][SS][DH];
    __shared__ int   valid[SS];
    int lane = threadIdx.x & 31;
    int head = threadIdx.x >> 5;
    int b = blockIdx.x / FF;
    int f = blockIdx.x % FF;

    if (threadIdx.x < SS)
        valid[threadIdx.x] = (threadIdx.x == HH) ? 1 : amask[b * HH + threadIdx.x];
    for (int i = lane; i < SS * DH; i += 32) {
        int t = i >> 4, e = i & 15;
        size_t idx = ((size_t)(b * SS + t) * FF + f) * DD + head * DH + e;
        kst[head][t][e] = bf2f(k[idx]);
        vst[head][t][e] = bf2f(v[idx]);
    }
    __syncthreads();

    if (lane < SS) {
        int st = lane;
        const u16* qr = q + ((size_t)(b * SS + st) * FF + f) * DD + head * DH;
        float qv[DH];
#pragma unroll
        for (int e = 0; e < DH; ++e) qv[e] = bf2f(qr[e]);
        float sc[SS], mx = -3.0e38f;
#pragma unroll
        for (int t = 0; t < SS; ++t) {
            float d = 0.f;
#pragma unroll
            for (int e = 0; e < DH; ++e) d += qv[e] * kst[head][t][e];
            d = valid[t] ? d * 0.25f : -1.0e9f;
            sc[t] = d; mx = fmaxf(mx, d);
        }
        float sum = 0.f;
#pragma unroll
        for (int t = 0; t < SS; ++t) { sc[t] = __expf(sc[t] - mx); sum += sc[t]; }
        float inv = 1.0f / sum;
        u16* orow = o + ((size_t)(b * SS + st) * FF + f) * DD + head * DH;
#pragma unroll
        for (int e = 0; e < DH; ++e) {
            float a = 0.f;
#pragma unroll
            for (int t = 0; t < SS; ++t) a += sc[t] * vst[head][t][e];
            orow[e] = f2bf(a * inv);
        }
    }
}

// ---------------- attention pooling over F ----------------
__global__ __launch_bounds__(128) void pool_kernel(const float* __restrict__ h,
                                                   const float* __restrict__ pw,
                                                   float* __restrict__ pooled) {
    __shared__ float lg[FF];
    int tid = threadIdx.x;
    const float* hb = h + (size_t)blockIdx.x * FF * DD;
    if (tid < FF) {
        float s = 0.f;
        for (int d = 0; d < DD; ++d) s += hb[tid * DD + d] * pw[d];
        lg[tid] = s;
    }
    __syncthreads();
    float mx = -3.0e38f;
#pragma unroll
    for (int f = 0; f < FF; ++f) mx = fmaxf(mx, lg[f]);
    float w[FF], se = 0.f;
#pragma unroll
    for (int f = 0; f < FF; ++f) { w[f] = __expf(lg[f] - mx); se += w[f]; }
    float inv = 1.0f / se;
    float acc = 0.f;
#pragma unroll
    for (int f = 0; f < FF; ++f) acc += w[f] * hb[f * DD + tid];
    pooled[(size_t)blockIdx.x * DD + tid] = acc * inv;
}

// ---------------- final projection: combined(256) @ proj_W(256x128) ----------------
__global__ __launch_bounds__(128) void proj_kernel(const float* __restrict__ pooled,
                                                   const float* __restrict__ pW,
                                                   const float* __restrict__ pb,
                                                   const int* __restrict__ amask,
                                                   float* __restrict__ out) {
    __shared__ float comb[2 * DD];
    int b  = blockIdx.x >> 4;
    int hh = blockIdx.x & 15;
    int n  = threadIdx.x;
    comb[n]      = pooled[((size_t)b * SS + hh) * DD + n];
    comb[DD + n] = pooled[((size_t)b * SS + HH) * DD + n];
    __syncthreads();
    float acc = pb[n];
    for (int j = 0; j < 2 * DD; ++j) acc += comb[j] * pW[j * DD + n];
    float m = (float)amask[b * HH + hh];
    out[(size_t)blockIdx.x * DD + n] = acc * m;
}

// ---------------- host orchestration ----------------
extern "C" void kernel_launch(void* const* d_in, const int* in_sizes, int n_in,
                              void* d_out, int out_size, void* d_ws, size_t ws_size,
                              hipStream_t stream) {
    (void)in_sizes; (void)n_in; (void)out_size; (void)ws_size;
    const float* x     = (const float*)d_in[0];
    const int*   amask = (const int*)d_in[1];
    auto P = [&](int i) { return (const float*)d_in[i]; };
    // param layout: per layer (base = 2 + 18*l):
    //  +0 ln1_g +1 ln1_b +2..5 fa{Wq,Wk,Wv,Wo} +6 ln2_g +7 ln2_b
    //  +8..11 sa{Wq,Wk,Wv,Wo} +12 ln3_g +13 ln3_b +14 W1 +15 b1 +16 W2 +17 b2
    // then: 38 cls, 39 proj_W, 40 proj_b, 41 pool_w
    const float* cls   = P(38);
    const float* projW = P(39);
    const float* projB = P(40);
    const float* poolW = P(41);

    char* ws = (char*)d_ws;
    const size_t bytes_h  = (size_t)NTOK * DD * 4;
    const size_t bytes_bf = (size_t)NTOK * DD * 2;
    const size_t off_h    = 0;
    const size_t off_ln   = off_h + bytes_h;
    const size_t off_R    = off_ln + bytes_bf;                 // union region NTOK*DFF*2
    const size_t off_pool = off_R + (size_t)NTOK * DFF * 2;
    const size_t off_wb   = off_pool + (size_t)BB * SS * DD * 4;

    float* h      = (float*)(ws + off_h);
    u16*   lnb    = (u16*)(ws + off_ln);
    u16*   qb     = (u16*)(ws + off_R);
    u16*   kb     = qb + (size_t)NTOK * DD;
    u16*   vb     = kb + (size_t)NTOK * DD;
    u16*   ob     = vb + (size_t)NTOK * DD;
    u16*   ffnb   = qb;                                        // reuses q/k/v/o region
    float* pooled = (float*)(ws + off_pool);
    u16*   wbf    = (u16*)(ws + off_wb);
    const size_t WLAYER = 8 * 16384 + 2 * 65536;               // bf16 elems per layer

    // ---- weight prep: fp32 -> bf16 transposed ----
    for (int l = 0; l < 2; ++l) {
        int base = 2 + l * 18;
        u16* wl = wbf + (size_t)l * WLAYER;
        for (int i = 0; i < 4; ++i)
            wprep_kernel<<<64, 256, 0, stream>>>(P(base + 2 + i), wl + i * 16384, DD, DD);
        for (int i = 0; i < 4; ++i)
            wprep_kernel<<<64, 256, 0, stream>>>(P(base + 8 + i), wl + 65536 + i * 16384, DD, DD);
        wprep_kernel<<<256, 256, 0, stream>>>(P(base + 14), wl + 131072, DD, DFF);
        wprep_kernel<<<256, 256, 0, stream>>>(P(base + 16), wl + 196608, DFF, DD);
    }

    // ---- h = concat(x, cls) ----
    concat_kernel<<<(NTOK * DD) / 256, 256, 0, stream>>>(x, cls, h);

    auto gemm = [&](const u16* A, const u16* Wt, const float* bias,
                    u16* oB, float* oF, int N, int K, int mode) {
        dim3 g(NTOK / 32, N / 128);
        gemm_wmma_kernel<<<g, 256, 0, stream>>>(A, Wt, bias, oB, oF, N, K, mode);
    };

    for (int l = 0; l < 2; ++l) {
        int base = 2 + l * 18;
        const u16* wl = wbf + (size_t)l * WLAYER;

        // ---- feature attention block ----
        ln_kernel<<<NTOK / 8, 256, 0, stream>>>(h, P(base + 0), P(base + 1), lnb, NTOK);
        gemm(lnb, wl + 0,     nullptr, qb, nullptr, DD, DD, 0);
        gemm(lnb, wl + 16384, nullptr, kb, nullptr, DD, DD, 0);
        gemm(lnb, wl + 32768, nullptr, vb, nullptr, DD, DD, 0);
        feat_attn_kernel<<<BB * SS, 256, 0, stream>>>(qb, kb, vb, ob);
        gemm(ob, wl + 49152, nullptr, nullptr, h, DD, DD, 2);   // h += o @ Wo

        // ---- sequence attention block ----
        ln_kernel<<<NTOK / 8, 256, 0, stream>>>(h, P(base + 6), P(base + 7), lnb, NTOK);
        gemm(lnb, wl + 65536, nullptr, qb, nullptr, DD, DD, 0);
        gemm(lnb, wl + 81920, nullptr, kb, nullptr, DD, DD, 0);
        gemm(lnb, wl + 98304, nullptr, vb, nullptr, DD, DD, 0);
        seq_attn_kernel<<<BB * FF, 256, 0, stream>>>(qb, kb, vb, amask, ob);
        gemm(ob, wl + 114688, nullptr, nullptr, h, DD, DD, 2);  // h += o @ Wo

        // ---- FFN ----
        ln_kernel<<<NTOK / 8, 256, 0, stream>>>(h, P(base + 12), P(base + 13), lnb, NTOK);
        gemm(lnb, wl + 131072, P(base + 15), ffnb, nullptr, DFF, DD, 1);  // gelu(z@W1+b1)
        gemm(ffnb, wl + 196608, P(base + 17), nullptr, h, DD, DFF, 2);    // h += .. @W2 + b2
    }

    pool_kernel<<<BB * SS, 128, 0, stream>>>(h, poolW, pooled);
    proj_kernel<<<BB * HH, 128, 0, stream>>>(pooled, projW, projB, amask, (float*)d_out);
}
